// PSAR_87574383166109
// MI455X (gfx1250) — matrix operand order repaired and checked by reading, not compiled, not run on hardware
//
#include <hip/hip_runtime.h>

typedef __bf16 bf16;
typedef __attribute__((ext_vector_type(16))) __bf16 v16bf;
typedef __attribute__((ext_vector_type(8)))  __bf16 v8bf;
typedef __attribute__((ext_vector_type(8)))  float  v8f;

// ---------------------------------------------------------------------------
// Weight pre-pack: [Cout][Cin][3][3] f32 -> WMMA B fragments, zero-padded.
// Layout: wp[nt][tap][ch][lane(32)][e(16)], lane = half*16+ncol:
//   value = W[nt*16+ncol][ch*32 + half*16 + e][tap]
// ---------------------------------------------------------------------------
__global__ void k_packw(const float* __restrict__ wgt, bf16* __restrict__ wp,
                        int Cin, int Cout) {
    const int chunks = (Cin + 31) >> 5;
    const int ntiles = (Cout + 15) >> 4;
    const int total = ntiles * 9 * chunks * 512;
    int i = blockIdx.x * blockDim.x + threadIdx.x;
    if (i >= total) return;
    const int e    = i & 15;
    const int lane = (i >> 4) & 31;
    int r = i >> 9;
    const int ch  = r % chunks; r /= chunks;
    const int tap = r % 9;
    const int nt  = r / 9;
    const int ncol = lane & 15, half = lane >> 4;
    const int cg = ch * 32 + half * 16 + e;
    const int co = nt * 16 + ncol;
    float v = (cg < Cin && co < Cout) ? wgt[((size_t)co * Cin + cg) * 9 + tap] : 0.f;
    wp[i] = (bf16)v;
}

// ---------------------------------------------------------------------------
// 3x3 SAME conv, implicit GEMM on v_wmma_f32_16x16x32_bf16.
// One wave -> 16(spatial row segment) x 16(cout) tile; K = 9 taps x 32-ch chunks.
// LDS halo tile per wave: [3 rows][18 cols][32 ch] bf16 (channel-last so the
// A fragment is two contiguous 8-channel runs -> 2x ds_load_b128 per tap).
// Tap loop software-pipelined: loads of tap t+1 issued before wmma of tap t.
// ---------------------------------------------------------------------------
__global__ __launch_bounds__(256) void k_conv3_wmma(
    const bf16* __restrict__ in, const v16bf* __restrict__ wp,
    const float* __restrict__ bias, const void* __restrict__ res,
    void* __restrict__ out,
    int N, int Cin, int Cout, int H, int W,
    int act, int ps2, int res_f32, int out_f32)
{
    __shared__ alignas(64) bf16 smem[8][3][18][32];   // 27.6 KB / block
    const int lane = threadIdx.x & 31;
    const int wv   = threadIdx.x >> 5;
    const int half = lane >> 4;
    const int ml   = lane & 15;

    const int tilesPerRow = W >> 4;
    const int tilesPerImg = (H * W) >> 4;
    const int tile = blockIdx.x * 8 + wv;
    const int n    = tile / tilesPerImg;
    const int rIn  = tile % tilesPerImg;
    const int y    = rIn / tilesPerRow;
    const int x0   = (rIn % tilesPerRow) << 4;

    const int chunks = (Cin + 31) >> 5;
    const v16bf* wpN = wp + (size_t)blockIdx.y * 9 * chunks * 32;
    const bf16* inN = in + (size_t)n * Cin * H * W;
    v8f acc = {};

    for (int ch = 0; ch < chunks; ++ch) {
        const int c0 = ch << 5;
        // ---- stage halo tile: unconditional clamped load + select-to-zero ----
        asm volatile("s_wait_dscnt 0" ::: "memory");
        for (int idx = lane; idx < 32 * 3 * 18; idx += 32) {
            const int ci  = idx / 54;
            const int rem = idx - ci * 54;
            const int row = rem / 18;
            const int col = rem - row * 18;
            const int ys = y + row - 1;
            const int xs = x0 + col - 1;
            const int cg = c0 + ci;
            const bool inb = (cg < Cin) & ((unsigned)ys < (unsigned)H)
                                        & ((unsigned)xs < (unsigned)W);
            const int cgC = cg < Cin ? cg : Cin - 1;
            const int ysC = ys < 0 ? 0 : (ys >= H ? H - 1 : ys);
            const int xsC = xs < 0 ? 0 : (xs >= W ? W - 1 : xs);
            const bf16 v = inN[((size_t)cgC * H + ysC) * W + xsC];
            smem[wv][row][col][ci] = inb ? v : (bf16)0.f;
        }
        asm volatile("s_wait_dscnt 0" ::: "memory");

        // ---- software-pipelined 9-tap WMMA sweep ----
        const v16bf* wpC = wpN + ch * 32 + lane;      // + tap*chunks*32 per tap
        const int bstride = chunks * 32;

        const bf16* ap0 = &smem[wv][0][ml + 0][0];    // tap 0: ky=0,kx=0
        v8bf lo = *(const v8bf*)(ap0 + half * 16);
        v8bf hi = *(const v8bf*)(ap0 + half * 16 + 8);
        v16bf bcur = wpC[0];

        #pragma unroll
        for (int tap = 0; tap < 9; ++tap) {
            const v16bf a = __builtin_shufflevector(lo, hi,
                    0,1,2,3,4,5,6,7,8,9,10,11,12,13,14,15);
            v8bf lo2 = lo, hi2 = hi;
            v16bf bnxt = bcur;
            if (tap < 8) {                            // prefetch tap+1 fragments
                const int kyn = (tap + 1) / 3, kxn = (tap + 1) % 3;
                const bf16* apn = &smem[wv][kyn][ml + kxn][0];
                lo2 = *(const v8bf*)(apn + half * 16);
                hi2 = *(const v8bf*)(apn + half * 16 + 8);
                bnxt = wpC[(tap + 1) * bstride];
            }
            acc = __builtin_amdgcn_wmma_f32_16x16x32_bf16(
                    false, a, false, bcur, (short)0, acc, false, false);
            lo = lo2; hi = hi2; bcur = bnxt;
        }
    }

    // ---- epilogue: C layout M = r + 8*half (spatial), N = lane%16 (cout) ----
    const int co = (blockIdx.y << 4) + ml;
    if (co < Cout) {
        const float bs = bias ? bias[co] : 0.f;
        float v[8];
        #pragma unroll
        for (int r = 0; r < 8; ++r) v[r] = acc[r] + bs;

        if (!ps2) {
            const size_t bidx = (((size_t)n * Cout + co) * H + y) * W + x0 + half * 8;
            if (res) {
                if (res_f32) {
                    #pragma unroll
                    for (int r = 0; r < 8; ++r) v[r] += ((const float*)res)[bidx + r];
                } else {
                    const v8bf rv = *(const v8bf*)((const bf16*)res + bidx);
                    #pragma unroll
                    for (int r = 0; r < 8; ++r) v[r] += (float)rv[r];
                }
            }
            #pragma unroll
            for (int r = 0; r < 8; ++r) {
                if      (act == 1) v[r] = v[r] > 0.f ? v[r] : 0.f;
                else if (act == 2) v[r] = v[r] >= 0.f ? v[r] : 0.1f * v[r];
                else if (act == 3) v[r] = 1.f / (1.f + __expf(-v[r]));
            }
            if (out_f32) {
                #pragma unroll
                for (int r = 0; r < 8; ++r) ((float*)out)[bidx + r] = v[r];
            } else {
                v8bf ov;
                #pragma unroll
                for (int r = 0; r < 8; ++r) ov[r] = (bf16)v[r];
                *(v8bf*)((bf16*)out + bidx) = ov;
            }
        } else {   // fused pixel-shuffle x2 (scattered stride-2 stores)
            #pragma unroll
            for (int r = 0; r < 8; ++r) {
                const int xo = x0 + r + half * 8;
                const int c = co >> 2, i2 = (co >> 1) & 1, j2 = co & 1;
                const size_t idx = (((size_t)n * (Cout >> 2) + c) * (size_t)(2 * H)
                                    + (2 * y + i2)) * (size_t)(2 * W) + (2 * xo + j2);
                float t = v[r];
                if      (act == 1) t = t > 0.f ? t : 0.f;
                else if (act == 2) t = t >= 0.f ? t : 0.1f * t;
                else if (act == 3) t = 1.f / (1.f + __expf(-t));
                if (out_f32) ((float*)out)[idx] = t;
                else         ((bf16*)out)[idx]  = (bf16)t;
            }
        }
    }
}

// ---------------------------------------------------------------------------
// Elementwise / helper kernels
// ---------------------------------------------------------------------------
__global__ void k_cvt(const float* __restrict__ in, bf16* __restrict__ out, int n) {
    int i = blockIdx.x * blockDim.x + threadIdx.x;
    if (i < n) out[i] = (bf16)in[i];
}

__global__ void k_patches(const float* __restrict__ x, bf16* __restrict__ xp) {
    int i = blockIdx.x * blockDim.x + threadIdx.x;
    if (i >= 144 * 3 * 1024) return;
    int xw = i & 31, yw = (i >> 5) & 31;
    int c = (i >> 10) % 3, p = i / 3072;
    int pj = p % 6, pi = (p / 6) % 6, b = p / 36;
    xp[i] = (bf16)x[((size_t)(b * 3 + c) * 192 + pi * 32 + yw) * 192 + pj * 32 + xw];
}

__global__ void k_std(const float* __restrict__ x, float* __restrict__ ssza) {
    __shared__ float s1[256], s2[256];
    int p = blockIdx.x;
    int pj = p % 6, pi = (p / 6) % 6, b = p / 36;
    float a = 0.f, q = 0.f;
    for (int e = threadIdx.x; e < 3072; e += 256) {
        int c = e >> 10, yw = (e >> 5) & 31, xw = e & 31;
        float v = x[((size_t)(b * 3 + c) * 192 + pi * 32 + yw) * 192 + pj * 32 + xw];
        a += v; q += v * v;
    }
    s1[threadIdx.x] = a; s2[threadIdx.x] = q;
    __syncthreads();
    for (int s = 128; s > 0; s >>= 1) {
        if (threadIdx.x < s) { s1[threadIdx.x] += s1[threadIdx.x + s];
                               s2[threadIdx.x] += s2[threadIdx.x + s]; }
        __syncthreads();
    }
    if (threadIdx.x == 0) {
        float m = s1[0] / 3072.f, m2 = s2[0] / 3072.f;
        float var = m2 - m * m;
        ssza[p] = sqrtf(var > 0.f ? var : 0.f);
    }
}

__global__ void k_bilinear(const float* __restrict__ x, float* __restrict__ base) {
    int i = blockIdx.x * blockDim.x + threadIdx.x;
    if (i >= 7077888) return;
    int ox = i % 768, oy = (i / 768) % 768;
    int c = (i / 589824) % 3, b = i / 1769472;
    float sy = (oy + 0.5f) * 0.25f - 0.5f;
    float sx = (ox + 0.5f) * 0.25f - 0.5f;
    sy = fminf(fmaxf(sy, 0.f), 191.f);
    sx = fminf(fmaxf(sx, 0.f), 191.f);
    int y0 = (int)sy, x0 = (int)sx;
    int y1 = min(y0 + 1, 191), x1 = min(x0 + 1, 191);
    float fy = sy - y0, fx = sx - x0;
    const float* xb = x + ((size_t)(b * 3 + c)) * 36864;
    float v00 = xb[y0 * 192 + x0], v01 = xb[y0 * 192 + x1];
    float v10 = xb[y1 * 192 + x0], v11 = xb[y1 * 192 + x1];
    base[i] = (v00 * (1 - fx) + v01 * fx) * (1 - fy) + (v10 * (1 - fx) + v11 * fx) * fy;
}

__global__ void k_route_t(const float* __restrict__ ssza, const float* tlp,
                          const float* trp, float* __restrict__ tvec) {
    int p = blockIdx.x * blockDim.x + threadIdx.x;
    if (p >= 144) return;
    float tl = *tlp, tr = *trp, thr = tl + tr, s = ssza[p];
    tvec[p] = (s > thr) ? 1e-4f * (s - thr)
            : (s >= tl) ? 2e-4f * tr
                        : 3e-4f * (tl - s);
}

__global__ void k_select(const bf16* __restrict__ xs, const bf16* __restrict__ xm,
                         const bf16* __restrict__ xh, const float* __restrict__ ssza,
                         const float* tlp, const float* trp, bf16* __restrict__ xsel) {
    int i = blockIdx.x * blockDim.x + threadIdx.x;
    if (i >= 144 * 64 * 1024) return;
    int p = i >> 16;
    float tl = *tlp, tr = *trp, thr = tl + tr, s = ssza[p];
    xsel[i] = (s > thr) ? xs[i] : (s >= tl) ? xm[i] : xh[i];
}

__global__ void k_combine(const bf16* __restrict__ xp, bf16* __restrict__ xo) {
    int i = blockIdx.x * blockDim.x + threadIdx.x;
    if (i >= 9437184) return;
    int xw = i % 192, yw = (i / 192) % 192;
    int c = (i / 36864) % 64, b = i / 2359296;
    int pi = yw / 32, yy = yw % 32, pj = xw / 32, xx = xw % 32;
    int n = (b * 6 + pi) * 6 + pj;
    xo[i] = xp[((size_t)(n * 64 + c) * 32 + yy) * 32 + xx];
}

__global__ void k_marm(const bf16* __restrict__ xo, const bf16* __restrict__ a,
                       bf16* __restrict__ y, int n) {
    int i = blockIdx.x * blockDim.x + threadIdx.x;
    if (i < n) y[i] = (bf16)((float)xo[i] * (1.f + (float)a[i]));
}

__global__ void k_timep(const float* __restrict__ tvec, float* __restrict__ out) {
    int b = threadIdx.x;
    if (b < 4) {
        float s = 0.f;
        for (int j = 0; j < 36; ++j) s += tvec[b * 36 + j];
        out[b] = s / 36.f;
    }
}

__global__ void k_fill1(float* __restrict__ p, int n) {
    int i = blockIdx.x * blockDim.x + threadIdx.x;
    if (i < n) p[i] = 1.f;
}

// ---------------------------------------------------------------------------
extern "C" void kernel_launch(void* const* d_in, const int* in_sizes, int n_in,
                              void* d_out, int out_size, void* d_ws, size_t ws_size,
                              hipStream_t stream) {
    const float* x        = (const float*)d_in[0];
    const float* w_iicm_f = (const float*)d_in[2];
    const float* b_iicm   = (const float*)d_in[3];
    const float* w_f1_f   = (const float*)d_in[4];
    const float* b_f1     = (const float*)d_in[5];
    const float* w_f2_f   = (const float*)d_in[6];
    const float* b_f2     = (const float*)d_in[7];
    const float* w_ts_f   = (const float*)d_in[8];
    const float* b_ts     = (const float*)d_in[9];
    const float* w_tm_f   = (const float*)d_in[10];
    const float* b_tm     = (const float*)d_in[11];
    const float* w_tk_f   = (const float*)d_in[12];
    const float* b_tk     = (const float*)d_in[13];
    const float* w_mid_f  = (const float*)d_in[14];
    const float* b_mid    = (const float*)d_in[15];
    const float* w_marm_f = (const float*)d_in[16];
    const float* b_marm   = (const float*)d_in[17];
    const float* w_u0_f   = (const float*)d_in[18];
    const float* b_u0     = (const float*)d_in[19];
    const float* w_u2_f   = (const float*)d_in[20];
    const float* b_u2     = (const float*)d_in[21];
    const float* w_u3_f   = (const float*)d_in[22];
    const float* b_u3     = (const float*)d_in[23];
    const float* w_u4_f   = (const float*)d_in[24];
    const float* b_u4     = (const float*)d_in[25];
    const float* tl       = (const float*)d_in[26];
    const float* tr       = (const float*)d_in[27];

    // ---- workspace bump allocator ----
    char*  ws  = (char*)d_ws;
    size_t off = 0;
    auto alloc = [&](size_t bytes) -> void* {
        void* p = ws + off;
        off = (off + bytes + 255) & ~(size_t)255;
        return p;
    };
    const size_t PB = (size_t)144 * 64 * 1024;

    bf16* w_iicm = (bf16*)alloc((size_t)4608 * 2);
    bf16* w_f1   = (bf16*)alloc((size_t)18432 * 2);
    bf16* w_f2   = (bf16*)alloc((size_t)36864 * 2);
    bf16* w_ts   = (bf16*)alloc((size_t)368640 * 2);
    bf16* w_tm   = (bf16*)alloc((size_t)368640 * 2);
    bf16* w_tk   = (bf16*)alloc((size_t)442368 * 2);
    bf16* w_mid  = (bf16*)alloc((size_t)36864 * 2);
    bf16* w_marm = (bf16*)alloc((size_t)18432 * 2);
    bf16* w_u0   = (bf16*)alloc((size_t)147456 * 2);
    bf16* w_u2   = (bf16*)alloc((size_t)147456 * 2);
    bf16* w_u3   = (bf16*)alloc((size_t)36864 * 2);
    bf16* w_u4   = (bf16*)alloc((size_t)9216 * 2);
    float* base  = (float*)alloc((size_t)7077888 * 4);
    bf16* xp     = (bf16*)alloc((size_t)442368 * 2);
    bf16* xfull  = (bf16*)alloc((size_t)442368 * 2);
    float* ssza  = (float*)alloc(144 * 4);
    float* tvec  = (float*)alloc(144 * 4);
    bf16* feat   = (bf16*)alloc((size_t)1769472 * 2);
    bf16* t1     = (bf16*)alloc(PB * 2);
    bf16* xcrop  = (bf16*)alloc(PB * 2);
    bf16* xs     = (bf16*)alloc(PB * 2);
    bf16* xm     = (bf16*)alloc(PB * 2);
    bf16* xh     = (bf16*)alloc(PB * 2);
    bf16* y1     = (bf16*)alloc((size_t)37748736 * 2);
    bf16* y2     = (bf16*)alloc((size_t)150994944 * 2);
    bf16* y3     = (bf16*)alloc((size_t)150994944 * 2);

    auto pack = [&](const float* src, bf16* dst, int Ci, int Co) {
        int n = ((Co + 15) / 16) * 9 * ((Ci + 31) / 32) * 512;
        k_packw<<<(n + 255) / 256, 256, 0, stream>>>(src, dst, Ci, Co);
    };
    auto conv = [&](const bf16* in, const bf16* w, const float* b, const void* res,
                    void* out, int N, int Ci, int Co, int H, int W,
                    int act, int ps2, int resf, int outf) {
        dim3 grid((unsigned)((size_t)N * H * W / 16 / 8), (unsigned)((Co + 15) / 16));
        k_conv3_wmma<<<grid, 256, 0, stream>>>(in, (const v16bf*)w, b, res, out,
                                               N, Ci, Co, H, W, act, ps2, resf, outf);
    };

    // pack all weights into WMMA fragment layout
    pack(w_iicm_f, w_iicm, 3, 12);
    pack(w_f1_f,   w_f1,  12, 64);
    pack(w_f2_f,   w_f2,  64, 64);
    for (int i = 0; i < 10; ++i)
        pack(w_ts_f + (size_t)i * 36864, w_ts + (size_t)i * 36864, 64, 64);
    for (int i = 0; i < 10; ++i)
        pack(w_tm_f + (size_t)i * 36864, w_tm + (size_t)i * 36864, 64, 64);
    for (int i = 0; i < 12; ++i)
        pack(w_tk_f + (size_t)i * 36864, w_tk + (size_t)i * 36864, 64, 64);
    pack(w_mid_f,  w_mid, 64, 64);
    pack(w_marm_f, w_marm, 3, 64);
    pack(w_u0_f,   w_u0,  64, 256);
    pack(w_u2_f,   w_u2,  64, 256);
    pack(w_u3_f,   w_u3,  64, 64);
    pack(w_u4_f,   w_u4,  64, 3);

    // preprocessing
    k_patches<<<1728, 256, 0, stream>>>(x, xp);
    k_cvt<<<(442368 + 255) / 256, 256, 0, stream>>>(x, xfull, 442368);
    k_std<<<144, 256, 0, stream>>>(x, ssza);
    k_bilinear<<<27648, 256, 0, stream>>>(x, base);

    // IICM + shallow features
    conv(xp,   w_iicm, b_iicm, nullptr, feat,  144, 3, 12, 32, 32, 2, 0, 0, 0);
    conv(feat, w_f1,   b_f1,   nullptr, t1,    144, 12, 64, 32, 32, 2, 0, 0, 0);
    conv(t1,   w_f2,   b_f2,   nullptr, xcrop, 144, 64, 64, 32, 32, 2, 0, 0, 0);

    // routed trunks: x = x + conv2(relu(conv1(x)))
    hipMemcpyAsync(xs, xcrop, PB * 2, hipMemcpyDeviceToDevice, stream);
    for (int i = 0; i < 5; ++i) {
        conv(xs, w_ts + (size_t)(i * 2 + 0) * 36864, b_ts + (i * 2 + 0) * 64,
             nullptr, t1, 144, 64, 64, 32, 32, 1, 0, 0, 0);
        conv(t1, w_ts + (size_t)(i * 2 + 1) * 36864, b_ts + (i * 2 + 1) * 64,
             xs, xs, 144, 64, 64, 32, 32, 0, 0, 0, 0);
    }
    hipMemcpyAsync(xm, xs, PB * 2, hipMemcpyDeviceToDevice, stream);
    for (int i = 0; i < 5; ++i) {
        conv(xm, w_tm + (size_t)(i * 2 + 0) * 36864, b_tm + (i * 2 + 0) * 64,
             nullptr, t1, 144, 64, 64, 32, 32, 1, 0, 0, 0);
        conv(t1, w_tm + (size_t)(i * 2 + 1) * 36864, b_tm + (i * 2 + 1) * 64,
             xm, xm, 144, 64, 64, 32, 32, 0, 0, 0, 0);
    }
    hipMemcpyAsync(xh, xm, PB * 2, hipMemcpyDeviceToDevice, stream);
    for (int i = 0; i < 6; ++i) {
        conv(xh, w_tk + (size_t)(i * 2 + 0) * 36864, b_tk + (i * 2 + 0) * 64,
             nullptr, t1, 144, 64, 64, 32, 32, 1, 0, 0, 0);
        conv(t1, w_tk + (size_t)(i * 2 + 1) * 36864, b_tk + (i * 2 + 1) * 64,
             xh, xh, 144, 64, 64, 32, 32, 0, 0, 0, 0);
    }

    // routing
    k_route_t<<<1, 160, 0, stream>>>(ssza, tl, tr, tvec);
    k_select<<<36864, 256, 0, stream>>>(xs, xm, xh, ssza, tl, tr, t1);

    // mid conv + xcrop residual, patch recombination
    conv(t1, w_mid, b_mid, xcrop, xs, 144, 64, 64, 32, 32, 0, 0, 0, 0);
    k_combine<<<36864, 256, 0, stream>>>(xs, xm);

    // MARM gate
    conv(xfull, w_marm, b_marm, nullptr, xh, 4, 3, 64, 192, 192, 3, 0, 0, 0);
    k_marm<<<36864, 256, 0, stream>>>(xm, xh, xcrop, (int)PB);

    // upsampling tower (PS2 fused)
    conv(xcrop, w_u0, b_u0, nullptr, y1, 4, 64, 256, 192, 192, 2, 1, 0, 0);
    conv(y1,    w_u2, b_u2, nullptr, y2, 4, 64, 256, 384, 384, 2, 1, 0, 0);
    conv(y2,    w_u3, b_u3, nullptr, y3, 4, 64, 64, 768, 768, 2, 0, 0, 0);

    // final conv + bilinear base residual -> f32 output
    float* out_f = (float*)d_out;
    conv(y3, w_u4, b_u4, base, out_f, 4, 64, 3, 768, 768, 0, 0, 1, 1);

    // timep, mask, duplicated masked output
    k_timep<<<1, 32, 0, stream>>>(tvec, out_f + 7077888);
    k_fill1<<<9216, 256, 0, stream>>>(out_f + 7077892, 2359296);
    hipMemcpyAsync(out_f + 9437188, out_f, (size_t)7077888 * 4,
                   hipMemcpyDeviceToDevice, stream);
}
